// MSAPairWeightedAveraging_23124103922383
// MI455X (gfx1250) — matrix operand order repaired and checked by reading, not compiled, not run on hardware
//
#include <hip/hip_runtime.h>
#include <cstdint>
#include <cstddef>

typedef __attribute__((ext_vector_type(16))) __bf16 v16bf;
typedef __attribute__((ext_vector_type(8)))  float  v8f;

__device__ __forceinline__ unsigned short f2bf(float f) {
    union { float f; uint32_t u; } v; v.f = f;
    uint32_t r = v.u + 0x7FFFu + ((v.u >> 16) & 1u);   // round-to-nearest-even
    return (unsigned short)(r >> 16);
}
__device__ __forceinline__ float bf2f(unsigned short s) {
    union { uint32_t u; float f; } v; v.u = ((uint32_t)s) << 16;
    return v.f;
}
__device__ __forceinline__ float fast_sigmoid(float x) {
    // v_exp + v_rcp (TRANS ops) instead of IEEE div sequence
    return __builtin_amdgcn_rcpf(1.f + __expf(-x));
}

union FragU {
    v16bf v;
    unsigned short s[16];
    uint4 q[2];
};

// ---------------------------------------------------------------------------
// Kernel 0: convert weight matrices to bf16 in workspace
// ---------------------------------------------------------------------------
__global__ void k_prep(const float* __restrict__ Wvg, const float* __restrict__ Wout,
                       unsigned short* __restrict__ wvg, unsigned short* __restrict__ wout) {
    int idx = blockIdx.x * 256 + threadIdx.x;
    if (idx < 64 * 512)  wvg[idx]  = f2bf(Wvg[idx]);
    if (idx < 256 * 64)  wout[idx] = f2bf(Wout[idx]);
}

// ---------------------------------------------------------------------------
// Kernel 1: LN(msa) @ W_vg -> values (bf16, [h][n][s][d]) + sigmoid gates
// (same layout). Block = 256 threads (8 waves), 16 rows of flattened (s,n).
// Tiles 0..1 cover output cols 0..255 (values), tiles 2..3 cover 256..511
// (gates) -> the values/gates split is compile-time, no lane divergence.
// ---------------------------------------------------------------------------
__global__ void k_vg(const float* __restrict__ msa,
                     const float* __restrict__ lng, const float* __restrict__ lnb,
                     const unsigned short* __restrict__ wvg,
                     unsigned short* __restrict__ values,
                     unsigned short* __restrict__ gates) {
    __shared__ __align__(16) unsigned short Atile[16][64];

    const int t   = threadIdx.x;
    const int row = t >> 4;
    const int c0  = (t & 15) * 4;
    const long R  = (long)blockIdx.x * 16 + row;      // flattened (s,n) row

    // ---- LayerNorm over 64 dims (16 threads / row, shfl-xor width 16) ----
    const float* xr = msa + R * 64 + c0;
    float x0 = xr[0], x1 = xr[1], x2 = xr[2], x3 = xr[3];
    float sum = x0 + x1 + x2 + x3;
    #pragma unroll
    for (int off = 8; off; off >>= 1) sum += __shfl_xor(sum, off, 16);
    float mean = sum * (1.f / 64.f);
    float d0 = x0 - mean, d1 = x1 - mean, d2 = x2 - mean, d3 = x3 - mean;
    float sq = d0 * d0 + d1 * d1 + d2 * d2 + d3 * d3;
    #pragma unroll
    for (int off = 8; off; off >>= 1) sq += __shfl_xor(sq, off, 16);
    float rstd = rsqrtf(sq * (1.f / 64.f) + 1e-5f);
    Atile[row][c0 + 0] = f2bf(d0 * rstd * lng[c0 + 0] + lnb[c0 + 0]);
    Atile[row][c0 + 1] = f2bf(d1 * rstd * lng[c0 + 1] + lnb[c0 + 1]);
    Atile[row][c0 + 2] = f2bf(d2 * rstd * lng[c0 + 2] + lnb[c0 + 2]);
    Atile[row][c0 + 3] = f2bf(d3 * rstd * lng[c0 + 3] + lnb[c0 + 3]);
    __syncthreads();

    // ---- GEMM 16x512, K=64: each wave does 4 N-tiles of 16 ----
    const int lane = t & 31, w = t >> 5;
    const int half = lane >> 4, mr = lane & 15;

    FragU a0, a1;   // A fragments (K=0..31 and K=32..63) from LDS
    a0.q[0] = *(const uint4*)&Atile[mr][      half * 8];
    a0.q[1] = *(const uint4*)&Atile[mr][16 +  half * 8];
    a1.q[0] = *(const uint4*)&Atile[mr][32 +  half * 8];
    a1.q[1] = *(const uint4*)&Atile[mr][48 +  half * 8];

    #pragma unroll
    for (int tile = 0; tile < 4; ++tile) {
        const int col0 = w * 16 + tile * 128;
        FragU b0, b1;
        #pragma unroll
        for (int s = 0; s < 16; ++s) {
          const int kk = (s < 8) ? half * 8 + s : 16 + half * 8 + (s - 8);
          b0.s[s] = wvg[(size_t)(0  + kk) * 512 + col0 + mr];
          b1.s[s] = wvg[(size_t)(32 + kk) * 512 + col0 + mr]; }
        v8f c = {};
        c = __builtin_amdgcn_wmma_f32_16x16x32_bf16(false, a0.v, false, b0.v,
                                                    (short)0, c, false, false);
        c = __builtin_amdgcn_wmma_f32_16x16x32_bf16(false, a1.v, false, b1.v,
                                                    (short)0, c, false, false);
        #pragma unroll
        for (int v = 0; v < 8; ++v) {
            const long Rg = (long)blockIdx.x * 16 + v + half * 8;
            const int si = (int)(Rg >> 9), nj = (int)(Rg & 511);
            if (tile < 2) {                             // values half (cols 0..255)
                const int col = col0 + mr;
                const int h = col >> 5, dd = col & 31;
                values[(((size_t)h * 512 + nj) * 256 + si) * 32 + dd] = f2bf(c[v]);
            } else {                                    // gates half (cols 256..511)
                const int cc = col0 + mr - 256;
                const int h = cc >> 5, dd = cc & 31;
                gates[(((size_t)h * 512 + nj) * 256 + si) * 32 + dd] =
                    f2bf(fast_sigmoid(c[v]));
            }
        }
    }
}

// ---------------------------------------------------------------------------
// Kernel 2: bias = LN(pair) @ W_pair ; softmax over j ; bf16 weights[h][i][j]
// One block per i (512 blocks), 256 threads.
// ---------------------------------------------------------------------------
__global__ void k_bias(const float* __restrict__ pair,
                       const float* __restrict__ lng, const float* __restrict__ lnb,
                       const float* __restrict__ Wp,
                       unsigned short* __restrict__ wts) {
    __shared__ float biasS[8][512];    // 16 KB
    __shared__ float WpS[128 * 8];     // 4 KB
    const int t = threadIdx.x;
    const int i = blockIdx.x;

    for (int k = t; k < 128 * 8; k += 256) WpS[k] = Wp[k];
    __syncthreads();

    #pragma unroll
    for (int jj = 0; jj < 2; ++jj) {
        const int j = t + jj * 256;
        const float* x = pair + ((size_t)i * 512 + j) * 128;
        float s = 0.f, ss = 0.f;
        for (int c = 0; c < 128; ++c) { float v = x[c]; s += v; ss += v * v; }
        float mean = s * (1.f / 128.f);
        float rstd = rsqrtf(ss * (1.f / 128.f) - mean * mean + 1e-5f);
        float acc[8] = {};
        for (int c = 0; c < 128; ++c) {
            float y = (x[c] - mean) * rstd * lng[c] + lnb[c];
            #pragma unroll
            for (int h = 0; h < 8; ++h) acc[h] += y * WpS[c * 8 + h];
        }
        #pragma unroll
        for (int h = 0; h < 8; ++h) biasS[h][j] = acc[h];
    }
    __syncthreads();

    // softmax over j: wave w owns head w (mask is all-true -> no-op)
    const int w = t >> 5, lane = t & 31;
    float mx = -3.4e38f;
    for (int j = lane; j < 512; j += 32) mx = fmaxf(mx, biasS[w][j]);
    #pragma unroll
    for (int off = 16; off; off >>= 1) mx = fmaxf(mx, __shfl_xor(mx, off, 32));
    float se = 0.f;
    for (int j = lane; j < 512; j += 32) se += __expf(biasS[w][j] - mx);
    #pragma unroll
    for (int off = 16; off; off >>= 1) se += __shfl_xor(se, off, 32);
    float inv = __builtin_amdgcn_rcpf(se);
    for (int j = lane; j < 512; j += 32)
        wts[((size_t)w * 512 + i) * 512 + j] = f2bf(__expf(biasS[w][j] - mx) * inv);
}

// ---------------------------------------------------------------------------
// Kernel 3: per-head GEMM weights[512x512] @ values[512x8192], gate fused.
// Block = 256 threads / 8 waves in a 2(M) x 4(N) arrangement; each wave owns
// a 16(M) x 64(N) output (4 accumulators): per K-step one A fragment feeds
// four WMMAs, and each lane loads B as scalar bf16s (per k row, 4 cols
// 16 apart). Block tile = 32(M) x 256(N).
// grid = (8192/256, 512/32, heads)
// ---------------------------------------------------------------------------
__global__ void k_attn(const unsigned short* __restrict__ wts,
                       const unsigned short* __restrict__ values,
                       const unsigned short* __restrict__ gates,
                       unsigned short* __restrict__ attn) {
    const int h = blockIdx.z;
    const int t = threadIdx.x, lane = t & 31, w = t >> 5;
    const int wm = w >> 2, wn = w & 3;
    const int row0 = blockIdx.y * 32 + wm * 16;       // i
    const int col0 = blockIdx.x * 256 + wn * 64;      // s*32+d (wave: 64 cols)
    const unsigned short* A = wts    + (size_t)h * 512 * 512;
    const unsigned short* V = values + (size_t)h * 512 * 8192;
    const unsigned short* G = gates  + (size_t)h * 512 * 8192;
    const int half = lane >> 4, mr = lane & 15;

    v8f c0 = {}, c1 = {}, c2 = {}, c3 = {};
    for (int k0 = 0; k0 < 512; k0 += 32) {
        FragU a, b0, b1, b2, b3;
        a.q[0] = *(const uint4*)(A + (size_t)(row0 + mr) * 512 + k0 +      half * 8);
        a.q[1] = *(const uint4*)(A + (size_t)(row0 + mr) * 512 + k0 + 16 + half * 8);
        #pragma unroll
        for (int s = 0; s < 16; ++s) {
          const int kk = (s < 8) ? half * 8 + s : 16 + half * 8 + (s - 8);
          const unsigned short* p = V + (size_t)(k0 + kk) * 8192 + col0 + mr;
          b0.s[s] = p[0];
          b1.s[s] = p[16];
          b2.s[s] = p[32];
          b3.s[s] = p[48]; }
        if (k0 + 32 < 512)
            __builtin_prefetch(V + (size_t)(k0 + 32 + lane) * 8192 + col0, 0, 1);
        c0 = __builtin_amdgcn_wmma_f32_16x16x32_bf16(false, a.v, false, b0.v,
                                                     (short)0, c0, false, false);
        c1 = __builtin_amdgcn_wmma_f32_16x16x32_bf16(false, a.v, false, b1.v,
                                                     (short)0, c1, false, false);
        c2 = __builtin_amdgcn_wmma_f32_16x16x32_bf16(false, a.v, false, b2.v,
                                                     (short)0, c2, false, false);
        c3 = __builtin_amdgcn_wmma_f32_16x16x32_bf16(false, a.v, false, b3.v,
                                                     (short)0, c3, false, false);
    }
    #pragma unroll
    for (int v = 0; v < 8; ++v) {
        const int i = row0 + v + half * 8;
        #pragma unroll
        for (int q = 0; q < 4; ++q) {
            const int n  = col0 + q * 16 + mr;
            const int si = n >> 5, dd = n & 31;
            const float cv = (q == 0) ? c0[v] : (q == 1) ? c1[v] : (q == 2) ? c2[v] : c3[v];
            float gate = bf2f(G[(size_t)i * 8192 + n]);
            attn[((size_t)si * 512 + i) * 256 + h * 32 + dd] = f2bf(cv * gate);
        }
    }
}

// ---------------------------------------------------------------------------
// Kernel 4: attn[131072x256] @ W_out[256x64] -> f32 out.
// Block = 128 threads / 4 waves, 16 rows per block, each wave one 16-col tile.
// ---------------------------------------------------------------------------
__global__ void k_out(const unsigned short* __restrict__ attn,
                      const unsigned short* __restrict__ wout,
                      float* __restrict__ out) {
    const int t = threadIdx.x, lane = t & 31, w = t >> 5;
    const long row0 = (long)blockIdx.x * 16;
    const int col0 = w * 16;
    const int half = lane >> 4, mr = lane & 15;

    v8f c = {};
    #pragma unroll
    for (int k0 = 0; k0 < 256; k0 += 32) {
        FragU a, b;
        a.q[0] = *(const uint4*)(attn + (size_t)(row0 + mr) * 256 + k0 +      half * 8);
        a.q[1] = *(const uint4*)(attn + (size_t)(row0 + mr) * 256 + k0 + 16 + half * 8);
        #pragma unroll
        for (int s = 0; s < 16; ++s) {
          const int kk = (s < 8) ? half * 8 + s : 16 + half * 8 + (s - 8);
          b.s[s] = wout[(size_t)(k0 + kk) * 64 + col0 + mr]; }
        c = __builtin_amdgcn_wmma_f32_16x16x32_bf16(false, a.v, false, b.v,
                                                    (short)0, c, false, false);
    }
    #pragma unroll
    for (int v = 0; v < 8; ++v)
        out[(size_t)(row0 + v + half * 8) * 64 + col0 + mr] = c[v];
}

// ---------------------------------------------------------------------------
extern "C" void kernel_launch(void* const* d_in, const int* in_sizes, int n_in,
                              void* d_out, int out_size, void* d_ws, size_t ws_size,
                              hipStream_t stream) {
    const float* msa   = (const float*)d_in[0];
    const float* pair  = (const float*)d_in[1];
    const float* lmg   = (const float*)d_in[2];
    const float* lmb   = (const float*)d_in[3];
    const float* Wvg   = (const float*)d_in[4];
    const float* lpg   = (const float*)d_in[5];
    const float* lpb   = (const float*)d_in[6];
    const float* Wp    = (const float*)d_in[7];
    const float* Wout  = (const float*)d_in[8];
    float* out = (float*)d_out;

    char* ws = (char*)d_ws;
    unsigned short* values = (unsigned short*)(ws);                 // 67,108,864 B
    unsigned short* gates  = (unsigned short*)(ws +  67108864);     // 67,108,864 B
    unsigned short* attn   = (unsigned short*)(ws + 134217728);     // 67,108,864 B
    unsigned short* wts    = (unsigned short*)(ws + 201326592);     //  4,194,304 B
    unsigned short* wvg    = (unsigned short*)(ws + 205520896);     //     65,536 B
    unsigned short* wout   = (unsigned short*)(ws + 205586432);     //     32,768 B

    k_prep<<<128, 256, 0, stream>>>(Wvg, Wout, wvg, wout);
    k_vg  <<<8192, 256, 0, stream>>>(msa, lmg, lmb, wvg, values, gates);
    k_bias<<<512, 256, 0, stream>>>(pair, lpg, lpb, Wp, wts);
    k_attn<<<dim3(32, 16, 8), 256, 0, stream>>>(wts, values, gates, attn);
    k_out <<<8192, 128, 0, stream>>>(attn, wout, out);
}
